// VisionTransformerMoE_5506148073590
// MI455X (gfx1250) — matrix operand order repaired and checked by reading, not compiled, not run on hardware
//
#include <hip/hip_runtime.h>
#include <hip/hip_bf16.h>
#include <math.h>

typedef __bf16 bf16_t;
typedef bf16_t v16bf __attribute__((ext_vector_type(16)));
typedef bf16_t v8bf  __attribute__((ext_vector_type(8)));
typedef float  v8f   __attribute__((ext_vector_type(8)));

#define TT    6272      // B*N tokens
#define DD    768
#define NH    12
#define HD    64
#define NSEQ  1568
#define BB    4
#define HM    192
#define CAPTY 3136      // T*K/E*1.0

// ---------------- fragment helpers (CDNA5 WMMA layouts) ----------------
// A-frag (16x32 bf16): lane m=lane&15; K elems {kb..kb+7, kb+16..kb+23}, kb=(lane<16)?0:8
static __device__ inline v16bf load_a_frag(const bf16_t* p) {
  v8bf lo = *(const v8bf*)p;
  v8bf hi = *(const v8bf*)(p + 16);
  return __builtin_shufflevector(lo, hi, 0,1,2,3,4,5,6,7,8,9,10,11,12,13,14,15);
}
// B-frag (32x16 bf16): lane n=lane&15; K elems {ks..ks+7, ks+16..ks+23}, ks=(lane<16)?0:8
static __device__ inline v16bf load_b_frag(const bf16_t* p) {
  v8bf lo = *(const v8bf*)p;
  v8bf hi = *(const v8bf*)(p + 16);
  return __builtin_shufflevector(lo, hi, 0,1,2,3,4,5,6,7,8,9,10,11,12,13,14,15);
}
static __device__ inline v8f wmma_bf16(v16bf a, v16bf b, v8f c) {
  return __builtin_amdgcn_wmma_f32_16x16x32_bf16(false, a, false, b, (short)0, c, false, false);
}
static __device__ inline float gelu_exact(float x) {
  return 0.5f * x * (1.0f + erff(x * 0.70710678118654752f));
}
static __device__ inline float rmax16(float v) {
  v = fmaxf(v, __shfl_xor(v, 1, 32));
  v = fmaxf(v, __shfl_xor(v, 2, 32));
  v = fmaxf(v, __shfl_xor(v, 4, 32));
  v = fmaxf(v, __shfl_xor(v, 8, 32));
  return v;
}
static __device__ inline float rsum16(float v) {
  v += __shfl_xor(v, 1, 32);
  v += __shfl_xor(v, 2, 32);
  v += __shfl_xor(v, 4, 32);
  v += __shfl_xor(v, 8, 32);
  return v;
}

// ---------------- weight transpose + bf16 convert: Wt[n*K+k] = W[k*N+n] ----------------
__global__ void wstage_k(const float* __restrict__ W, bf16_t* __restrict__ Wt, int K, int N) {
  int idx = blockIdx.x * 256 + threadIdx.x;
  if (idx >= K * N) return;
  int n = idx % N, k = idx / N;
  Wt[(size_t)n * K + k] = (bf16_t)W[idx];
}

// ---------------- layernorm: per-token row of 768 ----------------
__global__ void layernorm_k(const float* __restrict__ x, const float* __restrict__ g,
                            const float* __restrict__ bta,
                            float* __restrict__ outf, bf16_t* __restrict__ outb) {
  int t = blockIdx.x, tid = threadIdx.x;
  __shared__ float s1[256], s2[256];
  const float* row = x + (size_t)t * DD;
  float v0 = row[tid], v1 = row[tid + 256], v2 = row[tid + 512];
  s1[tid] = v0 + v1 + v2;
  s2[tid] = v0 * v0 + v1 * v1 + v2 * v2;
  __syncthreads();
  for (int s = 128; s > 0; s >>= 1) {
    if (tid < s) { s1[tid] += s1[tid + s]; s2[tid] += s2[tid + s]; }
    __syncthreads();
  }
  float mean = s1[0] * (1.0f / 768.0f);
  float var  = s2[0] * (1.0f / 768.0f) - mean * mean;
  float rstd = rsqrtf(var + 1e-5f);
  #pragma unroll
  for (int k = 0; k < 3; k++) {
    int d = tid + k * 256;
    float y = (row[d] - mean) * rstd * g[d] + bta[d];
    if (outf) outf[(size_t)t * DD + d] = y;
    if (outb) outb[(size_t)t * DD + d] = (bf16_t)y;
  }
}

// ---------------- generic WMMA GEMM: C[M,N] = A[M,K] @ W, W staged as Wt[N,K] bf16 ----------------
// Each wave computes a 32x64 tile (2 A-frags x 4 B-frags -> 8 accumulators):
// 12 b128 loads per 8 WMMAs instead of 10 per 4 (2x better operand reuse).
#define F_BIAS 1
#define F_GELU 2
#define F_RES  4
#define F_OUTF 8
#define F_OUTB 16
#define F_ACC  32
__global__ void gemm_bf16_k(const bf16_t* __restrict__ A, const bf16_t* __restrict__ Wt,
                            int M, int N, int K, int flags,
                            const float* __restrict__ bias, const float* __restrict__ res,
                            const float* __restrict__ rowscale, int rs_stride,
                            float* __restrict__ outf, bf16_t* __restrict__ outb) {
  int tid = threadIdx.x;
  int wv = tid >> 5, lane = tid & 31;
  int wid = blockIdx.x * 8 + wv;
  int tm_n = M >> 5, tn_n = N >> 6;
  if (wid >= tm_n * tn_n) return;
  int tm = wid % tm_n, tn = wid / tm_n;
  int row0 = tm << 5, col0 = tn << 6;
  int mrow = lane & 15;
  int kb = (lane < 16) ? 0 : 8;    // A-frag K base
  int ks = (lane < 16) ? 0 : 8;    // B-frag K base
  v8f c[2][4];
  #pragma unroll
  for (int i = 0; i < 2; i++)
    #pragma unroll
    for (int j = 0; j < 4; j++) c[i][j] = v8f{0};
  const bf16_t* arow0 = A + (size_t)(row0 + mrow) * K + kb;
  const bf16_t* arow1 = A + (size_t)(row0 + 16 + mrow) * K + kb;
  for (int k0 = 0; k0 < K; k0 += 32) {
    v16bf a0 = load_a_frag(arow0 + k0);
    v16bf a1 = load_a_frag(arow1 + k0);
    #pragma unroll
    for (int j = 0; j < 4; j++) {
      v16bf b = load_b_frag(Wt + (size_t)(col0 + j * 16 + mrow) * K + k0 + ks);
      c[0][j] = wmma_bf16(a0, b, c[0][j]);
      c[1][j] = wmma_bf16(a1, b, c[1][j]);
    }
  }
  #pragma unroll
  for (int i = 0; i < 2; i++) {
    int mg = row0 + i * 16 + ((lane < 16) ? 0 : 8);
    #pragma unroll
    for (int j = 0; j < 4; j++) {
      int n = col0 + j * 16 + mrow;
      float bval = (flags & F_BIAS) ? bias[n] : 0.0f;
      #pragma unroll
      for (int r = 0; r < 8; r++) {
        int m = mg + r;
        float v = c[i][j][r] + bval;
        if (flags & F_GELU) v = gelu_exact(v);
        size_t idx = (size_t)m * N + n;
        if (flags & F_ACC) {
          outf[idx] += rowscale[(size_t)m * rs_stride] * v;
        } else {
          float vo = v;
          if (flags & F_RES) vo += res[idx];
          if (flags & F_OUTF) outf[idx] = vo;
        }
        if (flags & F_OUTB) outb[idx] = (bf16_t)v;
      }
    }
  }
}

// ---------------- qkv staging: split/scale/transpose into per-head bf16 ----------------
__global__ void qkv_stage_k(const float* __restrict__ qkv, bf16_t* __restrict__ Qb,
                            bf16_t* __restrict__ Kb, bf16_t* __restrict__ Vt) {
  size_t idx = (size_t)blockIdx.x * 256 + threadIdx.x;
  if (idx >= (size_t)TT * 2304) return;
  int c = (int)(idx % 2304);
  size_t t = idx / 2304;
  int s = c / 768, hc = c % 768;
  int h = hc >> 6, d = hc & 63;
  int b = (int)(t / NSEQ), n = (int)(t % NSEQ);
  float v = qkv[idx];
  size_t bh = (size_t)(b * NH + h);
  if (s == 0)      Qb[(bh * NSEQ + n) * HD + d] = (bf16_t)(v * 0.125f);  // *hd^-0.5
  else if (s == 1) Kb[(bh * NSEQ + n) * HD + d] = (bf16_t)v;
  else             Vt[(bh * HD + d) * NSEQ + n] = (bf16_t)v;             // transposed [feat][tok]
}

// ---------------- flash attention: wave = 16 q-rows of one (b,h); WMMA QK^T and PV ----------------
__global__ void flash_attn_k(const bf16_t* __restrict__ Qb, const bf16_t* __restrict__ Kb,
                             const bf16_t* __restrict__ Vt, bf16_t* __restrict__ attn) {
  __shared__ __align__(16) bf16_t pbuf[8][16 * 32];
  int tid = threadIdx.x;
  int wv = tid >> 5, lane = tid & 31;
  int wid = blockIdx.x * 8 + wv;
  const int ntiles = NSEQ / 16;  // 98
  if (wid >= BB * NH * ntiles) return;
  int qt = wid % ntiles, bh = wid / ntiles;
  int h = bh % NH, b = bh / NH;
  const bf16_t* Q = Qb + (size_t)bh * NSEQ * HD;
  const bf16_t* K = Kb + (size_t)bh * NSEQ * HD;
  const bf16_t* V = Vt + (size_t)bh * HD * NSEQ;
  int mrow = lane & 15;
  int kb = (lane < 16) ? 0 : 8;
  int ks = (lane < 16) ? 0 : 8;
  int mloc = (lane < 16) ? 0 : 8;
  int q0 = qt * 16;
  v16bf qa0 = load_a_frag(Q + (size_t)(q0 + mrow) * HD + kb);
  v16bf qa1 = load_a_frag(Q + (size_t)(q0 + mrow) * HD + 32 + kb);
  v8f o[4] = {v8f{0}, v8f{0}, v8f{0}, v8f{0}};
  float mo[8], l[8];
  #pragma unroll
  for (int r = 0; r < 8; r++) { mo[r] = -INFINITY; l[r] = 0.0f; }
  bf16_t* pb = pbuf[wv];

  for (int k0 = 0; k0 < NSEQ; k0 += 32) {
    // S = Q @ K^T for 32 keys -> two 16x16 frags
    v8f s0 = v8f{0}, s1 = v8f{0};
    {
      v16bf bk = load_b_frag(K + (size_t)(k0 + mrow) * HD + ks);
      s0 = wmma_bf16(qa0, bk, s0);
      bk = load_b_frag(K + (size_t)(k0 + mrow) * HD + 32 + ks);
      s0 = wmma_bf16(qa1, bk, s0);
      bk = load_b_frag(K + (size_t)(k0 + 16 + mrow) * HD + ks);
      s1 = wmma_bf16(qa0, bk, s1);
      bk = load_b_frag(K + (size_t)(k0 + 16 + mrow) * HD + 32 + ks);
      s1 = wmma_bf16(qa1, bk, s1);
    }
    // online softmax (row = r + mloc, values spread across 16 lanes of half-wave)
    #pragma unroll
    for (int r = 0; r < 8; r++) {
      float tmax = rmax16(fmaxf(s0[r], s1[r]));
      float mn = fmaxf(mo[r], tmax);
      float a = __expf(mo[r] - mn);
      mo[r] = mn;
      s0[r] = __expf(s0[r] - mn);
      s1[r] = __expf(s1[r] - mn);
      float rs = rsum16(s0[r] + s1[r]);
      l[r] = l[r] * a + rs;
      o[0][r] *= a; o[1][r] *= a; o[2][r] *= a; o[3][r] *= a;
    }
    // relayout P (D-layout) -> A-layout via per-wave LDS
    #pragma unroll
    for (int r = 0; r < 8; r++) {
      pb[(mloc + r) * 32 + mrow]      = (bf16_t)s0[r];
      pb[(mloc + r) * 32 + 16 + mrow] = (bf16_t)s1[r];
    }
    asm volatile("s_wait_dscnt 0" ::: "memory");
    v16bf pa = load_a_frag(pb + mrow * 32 + kb);
    // O += P @ V  (V staged transposed: [feat][tok], so B-frag loads are contiguous)
    #pragma unroll
    for (int jf = 0; jf < 4; jf++) {
      v16bf bv = load_b_frag(V + (size_t)(jf * 16 + mrow) * NSEQ + k0 + ks);
      o[jf] = wmma_bf16(pa, bv, o[jf]);
    }
  }
  // normalize and store bf16 to attention staging [T, 768]
  #pragma unroll
  for (int r = 0; r < 8; r++) {
    float inv = 1.0f / l[r];
    size_t tok = (size_t)b * NSEQ + q0 + mloc + r;
    #pragma unroll
    for (int jf = 0; jf < 4; jf++) {
      attn[tok * DD + h * 64 + jf * 16 + mrow] = (bf16_t)(o[jf][r] * inv);
    }
  }
}

// ---------------- router: logits, noisy top-2, sparse softmax gate ----------------
__global__ void router_k(const float* __restrict__ x1, const float* __restrict__ noise,
                         const float* __restrict__ wr, const float* __restrict__ br,
                         const float* __restrict__ wn, const float* __restrict__ bn,
                         float* __restrict__ gatew, int* __restrict__ mask) {
  int t = blockIdx.x, tid = threadIdx.x;
  __shared__ float red[8][257];
  float acc[8] = {0, 0, 0, 0, 0, 0, 0, 0};
  const float* row = x1 + (size_t)t * DD;
  #pragma unroll
  for (int k = 0; k < 3; k++) {
    int d = tid + k * 256;
    float xv = row[d];
    #pragma unroll
    for (int e = 0; e < 4; e++) {
      acc[e]     += xv * wr[d * 4 + e];
      acc[4 + e] += xv * wn[d * 4 + e];
    }
  }
  #pragma unroll
  for (int e = 0; e < 8; e++) red[e][tid] = acc[e];
  __syncthreads();
  for (int s = 128; s > 0; s >>= 1) {
    if (tid < s) {
      #pragma unroll
      for (int e = 0; e < 8; e++) red[e][tid] += red[e][tid + s];
    }
    __syncthreads();
  }
  if (tid == 0) {
    float noisy[4];
    #pragma unroll
    for (int e = 0; e < 4; e++) {
      float lg = red[e][0] + br[e];
      float nl = red[4 + e][0] + bn[e];
      float sp = (nl > 20.0f) ? nl : log1pf(__expf(nl));  // softplus
      noisy[e] = lg + noise[(size_t)t * 4 + e] * sp;
    }
    int i0 = 0;
    for (int e = 1; e < 4; e++) if (noisy[e] > noisy[i0]) i0 = e;
    int i1 = -1;
    for (int e = 0; e < 4; e++) if (e != i0 && (i1 < 0 || noisy[e] > noisy[i1])) i1 = e;
    float mx = noisy[i0];
    float Z = 1.0f + __expf(noisy[i1] - mx);
    #pragma unroll
    for (int e = 0; e < 4; e++) {
      bool sel = (e == i0) || (e == i1);
      gatew[(size_t)t * 4 + e] = sel ? (__expf(noisy[e] - mx) / Z) : 0.0f;
      mask[(size_t)t * 4 + e] = sel ? 1 : 0;
    }
  }
}

// ---------------- capacity: sequential per-expert cumsum over tokens (1 workgroup) ----------------
__global__ void capacity_k(const int* __restrict__ mask, const float* __restrict__ gatew,
                           float* __restrict__ wout) {
  __shared__ int sc[256];
  int tid = threadIdx.x;
  int base[4] = {0, 0, 0, 0};
  for (int start = 0; start < TT; start += 256) {
    int t = start + tid;
    #pragma unroll
    for (int e = 0; e < 4; e++) {
      int m = (t < TT) ? mask[(size_t)t * 4 + e] : 0;
      sc[tid] = m;
      __syncthreads();
      for (int off = 1; off < 256; off <<= 1) {
        int v = sc[tid];
        int add = (tid >= off) ? sc[tid - off] : 0;
        __syncthreads();
        sc[tid] = v + add;
        __syncthreads();
      }
      int pos = base[e] + sc[tid];
      if (t < TT) wout[(size_t)t * 4 + e] = (m && pos <= CAPTY) ? gatew[(size_t)t * 4 + e] : 0.0f;
      base[e] += sc[255];
      __syncthreads();
    }
  }
}

// ======================= host orchestration =======================
extern "C" void kernel_launch(void* const* d_in, const int* in_sizes, int n_in,
                              void* d_out, int out_size, void* d_ws, size_t ws_size,
                              hipStream_t stream) {
  const float* x      = (const float*)d_in[0];
  const float* noise  = (const float*)d_in[1];
  const float* ln1_g  = (const float*)d_in[2];
  const float* ln1_b  = (const float*)d_in[3];
  const float* qkv_w  = (const float*)d_in[4];
  const float* proj_w = (const float*)d_in[5];
  const float* proj_b = (const float*)d_in[6];
  const float* ln2_g  = (const float*)d_in[7];
  const float* ln2_b  = (const float*)d_in[8];
  const float* wr     = (const float*)d_in[9];
  const float* br     = (const float*)d_in[10];
  const float* wn     = (const float*)d_in[11];
  const float* bn     = (const float*)d_in[12];
  const float* e_w1   = (const float*)d_in[13];
  const float* e_b1   = (const float*)d_in[14];
  const float* e_w2   = (const float*)d_in[15];
  const float* e_b2   = (const float*)d_in[16];
  const float* fc1_w  = (const float*)d_in[17];
  const float* fc1_b  = (const float*)d_in[18];
  const float* fc2_w  = (const float*)d_in[19];
  const float* fc2_b  = (const float*)d_in[20];
  float* out = (float*)d_out;

  char* base = (char*)d_ws;
  size_t off = 0;
  auto take = [&](size_t bytes) -> char* {
    char* p = base + off;
    off += (bytes + 255) & ~(size_t)255;
    return p;
  };
  const size_t BHND = (size_t)BB * NH * NSEQ * HD;  // per-head staging elems
  bf16_t* wtqkv = (bf16_t*)take((size_t)2304 * 768 * 2);
  bf16_t* wtproj = (bf16_t*)take((size_t)768 * 768 * 2);
  bf16_t* wtfc1 = (bf16_t*)take((size_t)192 * 768 * 2);
  bf16_t* wtfc2 = (bf16_t*)take((size_t)768 * 192 * 2);
  bf16_t* wte1 = (bf16_t*)take((size_t)4 * 192 * 768 * 2);
  bf16_t* wte2 = (bf16_t*)take((size_t)4 * 768 * 192 * 2);
  bf16_t* ln1bf = (bf16_t*)take((size_t)TT * DD * 2);
  bf16_t* Qb = (bf16_t*)take(BHND * 2);
  bf16_t* Kb = (bf16_t*)take(BHND * 2);
  bf16_t* Vt = (bf16_t*)take(BHND * 2);
  bf16_t* attnb = (bf16_t*)take((size_t)TT * DD * 2);
  bf16_t* x1b = (bf16_t*)take((size_t)TT * DD * 2);
  bf16_t* hb = (bf16_t*)take((size_t)TT * HM * 2);
  float* qkvf = (float*)take((size_t)TT * 2304 * 4);
  float* x2 = (float*)take((size_t)TT * DD * 4);
  float* x1f = (float*)take((size_t)TT * DD * 4);
  float* gatew = (float*)take((size_t)TT * 4 * 4);
  float* wfin = (float*)take((size_t)TT * 4 * 4);
  int* maskb = (int*)take((size_t)TT * 4 * 4);
  (void)ws_size; (void)n_in; (void)in_sizes; (void)out_size;

  auto cdiv = [](size_t a, size_t b) { return (int)((a + b - 1) / b); };
  auto gemm_blocks = [&](int M, int N) { return ((M / 32) * (N / 64) + 7) / 8; };

  // 0) stage weights (transpose + bf16)
  wstage_k<<<cdiv((size_t)768 * 2304, 256), 256, 0, stream>>>(qkv_w, wtqkv, 768, 2304);
  wstage_k<<<cdiv((size_t)768 * 768, 256), 256, 0, stream>>>(proj_w, wtproj, 768, 768);
  wstage_k<<<cdiv((size_t)768 * 192, 256), 256, 0, stream>>>(fc1_w, wtfc1, 768, 192);
  wstage_k<<<cdiv((size_t)192 * 768, 256), 256, 0, stream>>>(fc2_w, wtfc2, 192, 768);
  for (int e = 0; e < 4; e++) {
    wstage_k<<<cdiv((size_t)768 * 192, 256), 256, 0, stream>>>(
        e_w1 + (size_t)e * 768 * 192, wte1 + (size_t)e * 192 * 768, 768, 192);
    wstage_k<<<cdiv((size_t)192 * 768, 256), 256, 0, stream>>>(
        e_w2 + (size_t)e * 192 * 768, wte2 + (size_t)e * 768 * 192, 192, 768);
  }
  // 1) ln1 -> bf16
  layernorm_k<<<TT, 256, 0, stream>>>(x, ln1_g, ln1_b, nullptr, ln1bf);
  // 2) qkv GEMM (no bias)
  gemm_bf16_k<<<gemm_blocks(TT, 2304), 256, 0, stream>>>(
      ln1bf, wtqkv, TT, 2304, 768, F_OUTF, nullptr, nullptr, nullptr, 0, qkvf, nullptr);
  // 3) split/scale/transpose q,k,v
  qkv_stage_k<<<cdiv((size_t)TT * 2304, 256), 256, 0, stream>>>(qkvf, Qb, Kb, Vt);
  // 4) flash attention
  flash_attn_k<<<(BB * NH * (NSEQ / 16) + 7) / 8, 256, 0, stream>>>(Qb, Kb, Vt, attnb);
  // 5) proj + residual -> x2
  gemm_bf16_k<<<gemm_blocks(TT, 768), 256, 0, stream>>>(
      attnb, wtproj, TT, 768, 768, F_BIAS | F_RES | F_OUTF, proj_b, x, nullptr, 0, x2, nullptr);
  // 6) ln2 -> x1 (f32 for router, bf16 for GEMMs)
  layernorm_k<<<TT, 256, 0, stream>>>(x2, ln2_g, ln2_b, x1f, x1b);
  // 7) router + capacity
  router_k<<<TT, 256, 0, stream>>>(x1f, noise, wr, br, wn, bn, gatew, maskb);
  capacity_k<<<1, 256, 0, stream>>>(maskb, gatew, wfin);
  // 8) dense MLP: fc1(gelu) -> fc2(+bias + x2 residual) into d_out
  gemm_bf16_k<<<gemm_blocks(TT, 192), 256, 0, stream>>>(
      x1b, wtfc1, TT, 192, 768, F_BIAS | F_GELU | F_OUTB, fc1_b, nullptr, nullptr, 0, nullptr, hb);
  gemm_bf16_k<<<gemm_blocks(TT, 768), 256, 0, stream>>>(
      hb, wtfc2, TT, 768, 192, F_BIAS | F_RES | F_OUTF, fc2_b, x2, nullptr, 0, out, nullptr);
  // 9) experts: h_e = gelu(x1@w1+b1); out += w[t,e] * (h_e@w2 + b2)
  for (int e = 0; e < 4; e++) {
    gemm_bf16_k<<<gemm_blocks(TT, 192), 256, 0, stream>>>(
        x1b, wte1 + (size_t)e * 192 * 768, TT, 192, 768, F_BIAS | F_GELU | F_OUTB,
        e_b1 + (size_t)e * 192, nullptr, nullptr, 0, nullptr, hb);
    gemm_bf16_k<<<gemm_blocks(TT, 768), 256, 0, stream>>>(
        hb, wte2 + (size_t)e * 768 * 192, TT, 768, 192, F_BIAS | F_ACC,
        e_b2 + (size_t)e * 768, nullptr, wfin + e, 4, out, nullptr);
  }
}
